// SpikingConv3DLayer_separable_11381663334468
// MI455X (gfx1250) — matrix operand order repaired and checked by reading, not compiled, not run on hardware
//
#include <hip/hip_runtime.h>

typedef __attribute__((ext_vector_type(16))) _Float16 v16h;
typedef __attribute__((ext_vector_type(8)))  float    v8f;

#define NSTEP 22
#define NTOT  23068672ll   /* 16*64*22*32*32 */
#define SPIX  1048576.0f   /* 16*64*32*32   */

union H16 { v16h v; unsigned u[8]; };
union H2  { _Float16 h[2]; unsigned u; };

// ---------------------------------------------------------------------------
// Setup: d16 = -beta * d  (f16, 64x64), w16 = (1-beta)*w_pw (f16, 64x32),
// thr[co] = b[co]*(norm[co]+eps), tsum zeroed.
// Uses d[a,f] = sum_ci s[ci]*wpw[a,ci]*wpw[f,ci],  s[ci] = sum_k w_dw[ci,k]^2.
// ---------------------------------------------------------------------------
__global__ void setup_kernel(const float* __restrict__ w_dw, const float* __restrict__ w_pw,
                             const float* __restrict__ beta_p, const float* __restrict__ b_in,
                             _Float16* __restrict__ d16, _Float16* __restrict__ w16,
                             float* __restrict__ thr, float* __restrict__ tsum) {
  __shared__ float sArr[32];
  __shared__ float wp[64 * 32];
  int tid = threadIdx.x;
  if (tid < 32) {
    float s = 0.f;
    for (int k = 0; k < 27; ++k) { float v = w_dw[tid * 27 + k]; s += v * v; }
    sArr[tid] = s;
  }
  for (int i = tid; i < 64 * 32; i += 256) wp[i] = w_pw[i];
  if (tid < 32) tsum[tid] = 0.f;
  __syncthreads();
  float beta = beta_p[0];
  for (int i = tid; i < 64 * 64; i += 256) {
    int a = i >> 6, f = i & 63;
    float dv = 0.f;
    for (int ci = 0; ci < 32; ++ci) dv += sArr[ci] * wp[a * 32 + ci] * wp[f * 32 + ci];
    d16[i] = (_Float16)(-beta * dv);
    if (a == f) thr[a] = b_in[a] * (dv + 1e-8f);
  }
  float om = 1.0f - beta;
  for (int i = tid; i < 64 * 32; i += 256) w16[i] = (_Float16)(om * w_pw[i]);
}

// ---------------------------------------------------------------------------
// Depthwise 3x3x3 conv, output channels-last f16:  dw16[b,t,h,w,ci]
// block: 256 threads = 32 w x 8 ci ; grid = 16*20*32*4
// ---------------------------------------------------------------------------
__global__ void dw_kernel(const float* __restrict__ x, const float* __restrict__ w_dw,
                          _Float16* __restrict__ dw16) {
  int q = blockIdx.x;
  int cg = q & 3;  q >>= 2;
  int h  = q & 31; q >>= 5;
  int t  = q % 20;
  int b  = q / 20;
  int tid = threadIdx.x;
  int w  = tid & 31;
  int ci = (cg << 3) + (tid >> 5);
  const float* wt = w_dw + ci * 27;
  const float* xb = x + (size_t)(b * 32 + ci) * (20 * 1024);
  float acc = 0.f;
#pragma unroll
  for (int kt = 0; kt < 3; ++kt) {
    int tt = t + kt - 1; if (tt < 0 || tt >= 20) continue;
#pragma unroll
    for (int kh = 0; kh < 3; ++kh) {
      int hh = h + kh - 1; if (hh < 0 || hh >= 32) continue;
#pragma unroll
      for (int kw = 0; kw < 3; ++kw) {
        int ww = w + kw - 1; if (ww < 0 || ww >= 32) continue;
        acc += wt[kt * 9 + kh * 3 + kw] * xb[(size_t)tt * 1024 + hh * 32 + ww];
      }
    }
  }
  dw16[((size_t)((b * 20 + t) * 32 + h) * 32 + w) * 32 + ci] = (_Float16)acc;
}

// ---------------------------------------------------------------------------
// Fused pointwise conv + 22-step spiking recurrence.
// Wave = 16 pixels (fixed b,h; w0..w0+15) x all 64 channels.
// D/C tiles: M=channel(4 tiles of 16), N=pixel(16). 12 WMMAs / wave / step.
// ---------------------------------------------------------------------------
__global__ void __launch_bounds__(256)
recur_kernel(const _Float16* __restrict__ dw16, const _Float16* __restrict__ d16,
             const _Float16* __restrict__ w16, const float* __restrict__ thr,
             const float* __restrict__ beta_p, float* __restrict__ out,
             float* __restrict__ tsum) {
  __shared__ unsigned ldsB[8][2][8][32];   // [wave][k-chunk][vgpr j][lane]
  const int tid  = threadIdx.x;
  const int wv   = tid >> 5;
  const int lane = tid & 31;
  const int pix  = lane & 15;
  const int hiL  = lane >> 4;
  const int wid  = blockIdx.x * 8 + wv;    // 1024 waves total
  const int b    = wid >> 6;
  const int rem  = wid & 63;
  const int h    = rem >> 1;
  const int w0   = (rem & 1) << 4;
  const float beta = beta_p[0];

  const unsigned* d16u = (const unsigned*)d16;
  const unsigned* w16u = (const unsigned*)w16;

  // A operands: rows = out-channel (M = lane&15 within tile), 16-bit A 16x32 layout:
  // contiguous K (this code's order, B uses the same; not the ISA table order): lane-lo K 0..15, lane-hi K 16..31.
  H16 Ad[4][2], Aw[4];
#pragma unroll
  for (int mt = 0; mt < 4; ++mt) {
    int row = mt * 16 + pix;
#pragma unroll
    for (int v = 0; v < 8; ++v) {
      int kIdx = 2 * v + hiL * 16;   // even
      Ad[mt][0].u[v] = d16u[row * 32 + 0  + (kIdx >> 1)];
      Ad[mt][1].u[v] = d16u[row * 32 + 16 + (kIdx >> 1)];
      Aw[mt].u[v]    = w16u[row * 16 + (kIdx >> 1)];
    }
  }

  float thrv[4][8];
#pragma unroll
  for (int mt = 0; mt < 4; ++mt)
#pragma unroll
    for (int r = 0; r < 8; ++r)
      thrv[mt][r] = thr[mt * 16 + hiL * 8 + r];  // D layout: co = mt*16 + hiL*8 + r

  v8f mem[4] = {};
  H16 sb[2];
#pragma unroll
  for (int j = 0; j < 8; ++j) { sb[0].u[j] = 0u; sb[1].u[j] = 0u; }

  // out[b,co,t,h,w]: per-lane base for (mt=0, r=0)
  const long Pb = (long)(b * 64 + hiL * 8) * 22528 + h * 32 + w0 + pix;
  const bool vp = (w0 + pix) >= 1;          // pointwise pad in w

  for (int t = 0; t < NSTEP; ++t) {
    const bool hasInp = (t >= 1) && (t <= 20) && (h >= 1);  // wave-uniform

    // B operand for pointwise: K=ci (32) x N=pixel(16); channels-last f16 => 2x b128
    H16 Bdw;
    if (hasInp && vp) {
      size_t e = ((size_t)(((b * 20 + (t - 1)) * 32 + (h - 1)) * 32 + (w0 + pix - 1))) * 32
               + (size_t)hiL * 16;
      const uint4* p = (const uint4*)(dw16 + e);
      uint4 q0 = p[0], q1 = p[1];
      Bdw.u[0] = q0.x; Bdw.u[1] = q0.y; Bdw.u[2] = q0.z; Bdw.u[3] = q0.w;
      Bdw.u[4] = q1.x; Bdw.u[5] = q1.y; Bdw.u[6] = q1.z; Bdw.u[7] = q1.w;
    } else {
#pragma unroll
      for (int j = 0; j < 8; ++j) Bdw.u[j] = 0u;
    }

#pragma unroll
    for (int mt = 0; mt < 4; ++mt) {
      v8f c;
#pragma unroll
      for (int r = 0; r < 8; ++r) c[r] = mem[mt][r] * beta;          // mem*beta
      if (hasInp)                                                    // + (1-beta)*pointwise
        c = __builtin_amdgcn_wmma_f32_16x16x32_f16(false, Aw[mt].v, false, Bdw.v,
                                                   (short)0, c, false, false);
      // - beta * (spk . d)   (d pre-negated & symmetric)
      c = __builtin_amdgcn_wmma_f32_16x16x32_f16(false, Ad[mt][0].v, false, sb[0].v,
                                                 (short)0, c, false, false);
      c = __builtin_amdgcn_wmma_f32_16x16x32_f16(false, Ad[mt][1].v, false, sb[1].v,
                                                 (short)0, c, false, false);
      mem[mt] = c;
    }

    // Spike, store spk_rec, repack as next-step B operand via wave-private LDS.
    float cnt = 0.f;
#pragma unroll
    for (int mt = 0; mt < 4; ++mt) {
#pragma unroll
      for (int rp = 0; rp < 4; ++rp) {
        float s0 = mem[mt][2 * rp]     > thrv[mt][2 * rp]     ? 1.f : 0.f;
        float s1 = mem[mt][2 * rp + 1] > thrv[mt][2 * rp + 1] ? 1.f : 0.f;
        out[Pb + (long)(mt * 16 + 2 * rp)     * 22528 + t * 1024] = s0;
        out[Pb + (long)(mt * 16 + 2 * rp + 1) * 22528 + t * 1024] = s1;
        cnt += s0 + s1;
        H2 pk; pk.h[0] = (_Float16)s0; pk.h[1] = (_Float16)s1;
        ldsB[wv][mt >> 1][hiL * 4 + rp][((mt & 1) << 4) + pix] = pk.u;
      }
    }
#pragma unroll
    for (int c2 = 0; c2 < 2; ++c2)
#pragma unroll
      for (int j = 0; j < 8; ++j)
        sb[c2].u[j] = ldsB[wv][c2][j][lane];

    // per-step spike count for the losses
#pragma unroll
    for (int off = 16; off >= 1; off >>= 1) cnt += __shfl_xor(cnt, off, 32);
    if (lane == 0) atomicAdd(&tsum[t], cnt);
  }
}

__global__ void finalize_kernel(const float* __restrict__ tsum, float* __restrict__ out) {
  if (threadIdx.x == 0) {
    float tot = 0.f, mx = 0.f;
    for (int i = 0; i < NSTEP; ++i) { float v = tsum[i]; tot += v; mx = fmaxf(mx, v); }
    out[NTOT]     = 0.5f * tot / (float)NTOT;
    out[NTOT + 1] = mx / SPIX;
  }
}

extern "C" void kernel_launch(void* const* d_in, const int* in_sizes, int n_in,
                              void* d_out, int out_size, void* d_ws, size_t ws_size,
                              hipStream_t stream) {
  const float* x    = (const float*)d_in[0];
  const float* w_dw = (const float*)d_in[1];
  const float* w_pw = (const float*)d_in[2];
  const float* beta = (const float*)d_in[3];
  const float* b_in = (const float*)d_in[4];
  float* out = (float*)d_out;
  char* ws = (char*)d_ws;
  _Float16* dw16 = (_Float16*)ws;                 // 16*20*32*32*32 f16 = 20,971,520 B
  _Float16* d16  = (_Float16*)(ws + 20971520);    // 8 KB
  _Float16* w16  = (_Float16*)(ws + 20979712);    // 4 KB
  float*    thr  = (float*)   (ws + 20983808);    // 256 B
  float*    tsum = (float*)   (ws + 20984064);    // 128 B

  hipLaunchKernelGGL(setup_kernel, dim3(1), dim3(256), 0, stream,
                     w_dw, w_pw, beta, b_in, d16, w16, thr, tsum);
  hipLaunchKernelGGL(dw_kernel, dim3(40960), dim3(256), 0, stream, x, w_dw, dw16);
  hipLaunchKernelGGL(recur_kernel, dim3(128), dim3(256), 0, stream,
                     dw16, d16, w16, thr, beta, out, tsum);
  hipLaunchKernelGGL(finalize_kernel, dim3(1), dim3(32), 0, stream, tsum, out);
}
